// GNN_node_22668837388513
// MI455X (gfx1250) — matrix operand order repaired and checked by reading, not compiled, not run on hardware
//
#include <hip/hip_runtime.h>
#include <hip/hip_bf16.h>

typedef __attribute__((ext_vector_type(16))) _Float16 v16h;
typedef __attribute__((ext_vector_type(8)))  float    v8f;

struct alignas(16) H8 { _Float16 v[8]; };

__device__ __forceinline__ v16h load_frag(const _Float16* p0, const _Float16* p1) {
    H8 a = *(const H8*)p0;
    H8 b = *(const H8*)p1;
    v16h f;
#pragma unroll
    for (int i = 0; i < 8; ++i) { f[i] = a.v[i]; f[8 + i] = b.v[i]; }
    return f;
}

// ---------------------------------------------------------------------------
// Utility kernels
// ---------------------------------------------------------------------------
__global__ void zero_f32(float* __restrict__ p, int n) {
    int i = blockIdx.x * blockDim.x + threadIdx.x;
    if (i < n) p[i] = 0.0f;
}

// W [K][Ncols] row-major fp32 -> WT [Ncols][K] f16
__global__ void convert_wT(const float* __restrict__ W, _Float16* __restrict__ WT,
                           int K, int Ncols) {
    int idx = blockIdx.x * blockDim.x + threadIdx.x;
    if (idx >= K * Ncols) return;
    int k = idx / Ncols, n = idx % Ncols;
    WT[n * K + k] = (_Float16)W[idx];
}

// ---------------------------------------------------------------------------
// Node encoder: h = type_emb[x0] + attr_emb[x1] + depth_emb[clip(depth)]
// one wave per node, 4 channels per lane
// ---------------------------------------------------------------------------
__global__ void node_encode(const int* __restrict__ x, const int* __restrict__ depth,
                            const float* __restrict__ te, const float* __restrict__ ae,
                            const float* __restrict__ de, float* __restrict__ h, int N) {
    int tid = blockIdx.x * blockDim.x + threadIdx.x;
    int i = tid >> 5, c0 = (tid & 31) << 2;
    if (i >= N) return;
    int t = x[2 * i], a = x[2 * i + 1];
    int d = depth[i]; d = d < 0 ? 0 : (d > 20 ? 20 : d);
    float4 vt = *(const float4*)(te + t * 128 + c0);
    float4 va = *(const float4*)(ae + a * 128 + c0);
    float4 vd = *(const float4*)(de + d * 128 + c0);
    float4 o;
    o.x = vt.x + va.x + vd.x; o.y = vt.y + va.y + vd.y;
    o.z = vt.z + va.z + vd.z; o.w = vt.w + va.w + vd.w;
    *(float4*)(h + (size_t)i * 128 + c0) = o;
}

// ---------------------------------------------------------------------------
// Edge scatter: msg = relu(h[src] + edge_attr@We + be); agg[dst] += msg
// one wave per edge, 4 channels per lane; native f32 atomics (L2-resident)
// ---------------------------------------------------------------------------
__global__ void edge_scatter(const float* __restrict__ h, const int* __restrict__ ei,
                             const float* __restrict__ ea, const float* __restrict__ We,
                             const float* __restrict__ be, float* __restrict__ agg, int E) {
    int tid = blockIdx.x * blockDim.x + threadIdx.x;
    int e = tid >> 5, c0 = (tid & 31) << 2;
    if (e >= E) return;
    int src = ei[e];
    int dst = ei[E + e];
    float a0 = ea[2 * e], a1 = ea[2 * e + 1];
    float4 w0 = *(const float4*)(We + c0);
    float4 w1 = *(const float4*)(We + 128 + c0);
    float4 bb = *(const float4*)(be + c0);
    float4 hh = *(const float4*)(h + (size_t)src * 128 + c0);
    float m0 = fmaxf(hh.x + a0 * w0.x + a1 * w1.x + bb.x, 0.0f);
    float m1 = fmaxf(hh.y + a0 * w0.y + a1 * w1.y + bb.y, 0.0f);
    float m2 = fmaxf(hh.z + a0 * w0.z + a1 * w1.z + bb.z, 0.0f);
    float m3 = fmaxf(hh.w + a0 * w0.w + a1 * w1.w + bb.w, 0.0f);
    float* p = agg + (size_t)dst * 128 + c0;
    unsafeAtomicAdd(p + 0, m0);
    unsafeAtomicAdd(p + 1, m1);
    unsafeAtomicAdd(p + 2, m2);
    unsafeAtomicAdd(p + 3, m3);
}

// ---------------------------------------------------------------------------
// WMMA GEMM, fused A-transform + bias + BN-stats epilogue.
// MODE 1: A = (1+eps)*h + agg                      (GIN combine)
// MODE 2: A = relu(t * ab[k] + ab[KDIM+k])         (BN + ReLU of inner MLP)
// C[64 x 128] block per 128-thread WG; K chunked by 64 through LDS.
// All 8 B-fragments are preloaded per k-step so the 8 WMMAs issue
// back-to-back (one s_wait_dscnt instead of eight).
// ---------------------------------------------------------------------------
template <int MODE, int KDIM, int NCOLS>
__global__ __launch_bounds__(128) void gemm_wmma(
    const float* __restrict__ A0, const float* __restrict__ A1,
    const float* __restrict__ epsp, const _Float16* __restrict__ WT,
    const float* __restrict__ bias, float* __restrict__ out,
    float* __restrict__ colsum, float* __restrict__ colsumsq, int nrows) {
    constexpr int LDA = 72;                 // 64 K + 8 pad (f16) -> conflict-free
    __shared__ _Float16 sA[64 * LDA];
    __shared__ _Float16 sB[128 * LDA];

    const int tid  = threadIdx.x;
    const int wave = tid >> 5;
    const int lane = tid & 31;
    const int half = lane >> 4;
    const int mr   = lane & 15;
    const int rowBlock = blockIdx.x * 64;
    const int colBase  = blockIdx.y * 128;
    const float epsv = (MODE == 1) ? (1.0f + epsp[0]) : 0.0f;

    v8f zero = {};
    v8f acc[8];
#pragma unroll
    for (int t = 0; t < 8; ++t) acc[t] = zero;

    for (int kb = 0; kb < KDIM; kb += 64) {
        // ---- prefetch next K-chunk while this one is processed ----
        if (kb + 64 < KDIM) {
            __builtin_prefetch(WT + (size_t)(colBase + tid) * KDIM + kb + 64, 0, 0);
            int pr = rowBlock + (tid >> 1);
            if (pr < nrows)
                __builtin_prefetch(A0 + (size_t)pr * KDIM + kb + 64 + ((tid & 1) << 5), 0, 0);
        }
        // ---- cooperative A-chunk load: fp32 -> transform -> f16 LDS ----
        for (int idx = tid; idx < 64 * 64; idx += 128) {
            int r = idx >> 6, k = idx & 63;
            int gr = rowBlock + r, gk = kb + k;
            float v = 0.0f;
            if (gr < nrows) {
                if (MODE == 1) {
                    size_t o = (size_t)gr * KDIM + gk;
                    v = epsv * A0[o] + A1[o];
                } else {
                    float xv = A0[(size_t)gr * KDIM + gk];
                    v = fmaxf(xv * A1[gk] + A1[KDIM + gk], 0.0f);
                }
            }
            sA[r * LDA + k] = (_Float16)v;
        }
        // ---- cooperative B-chunk load: WT[colBase+n][kb..kb+64) ----
        for (int idx = tid; idx < 128 * 16; idx += 128) {
            int n = idx >> 4, kc = (idx & 15) << 2;
            const _Float16* src = WT + (size_t)(colBase + n) * KDIM + kb + kc;
            *(unsigned long long*)(sB + n * LDA + kc) =
                *(const unsigned long long*)src;
        }
        __syncthreads();

#pragma unroll
        for (int ks = 0; ks < 2; ++ks) {
            // A fragment: 16x32 f16 (K0-7/K16-23 per half-wave per ISA layout)
            const int aOff0 = ks * 32 + half * 8;
            const int aOff1 = aOff0 + 16;
            const _Float16* ap = sA + (wave * 16 + mr) * LDA;
            v16h aF = load_frag(ap + aOff0, ap + aOff1);
            // B fragments: 32x16 f16 (K0-7/K16-23 per half-wave, matching A)
            const int bOff0 = ks * 32 + half * 8;
            const int bOff1 = bOff0 + 16;
            v16h bFr[8];
#pragma unroll
            for (int t = 0; t < 8; ++t) {
                const _Float16* bp = sB + (t * 16 + mr) * LDA;
                bFr[t] = load_frag(bp + bOff0, bp + bOff1);
            }
#pragma unroll
            for (int t = 0; t < 8; ++t)
                acc[t] = __builtin_amdgcn_wmma_f32_16x16x32_f16(
                    false, aF, false, bFr[t], (short)0, acc[t], false, false);
        }
        __syncthreads();
    }

    // ---- epilogue: +bias, store, per-column BN stats (sum, sumsq) ----
    const int rowWave = rowBlock + wave * 16;
#pragma unroll
    for (int t = 0; t < 8; ++t) {
        int col = colBase + t * 16 + mr;
        float bv = bias[col];
        float s = 0.0f, s2 = 0.0f;
#pragma unroll
        for (int i = 0; i < 8; ++i) {
            int r = rowWave + half * 8 + i;       // C layout: VGPR i, half-wave M split
            float v = acc[t][i] + bv;
            if (r < nrows) {
                out[(size_t)r * NCOLS + col] = v;
                s += v; s2 += v * v;
            }
        }
        s  += __shfl_xor(s, 16, 32);
        s2 += __shfl_xor(s2, 16, 32);
        if (half == 0) {
            unsafeAtomicAdd(&colsum[col], s);
            unsafeAtomicAdd(&colsumsq[col], s2);
        }
    }
}

// ---------------------------------------------------------------------------
// BN finalize: fold (mu, var, gamma, beta) into y = a*x + b
// ---------------------------------------------------------------------------
__global__ void bn_finalize(const float* __restrict__ cs, const float* __restrict__ cq,
                            const float* __restrict__ g, const float* __restrict__ beta,
                            float* __restrict__ ab, int C, float invN) {
    int c = blockIdx.x * blockDim.x + threadIdx.x;
    if (c >= C) return;
    float mu  = cs[c] * invN;
    float var = cq[c] * invN - mu * mu;
    float a = g[c] * rsqrtf(var + 1e-5f);
    ab[c] = a;
    ab[C + c] = beta[c] - mu * a;
}

// outer BN apply (+ optional ReLU), writes next h (or final output)
__global__ void bn_apply(const float* __restrict__ u, const float* __restrict__ ab,
                         float* __restrict__ dst, int total, int C, int relu) {
    int idx = blockIdx.x * blockDim.x + threadIdx.x;
    if (idx >= total) return;
    int c = idx & (C - 1);
    float v = u[idx] * ab[c] + ab[C + c];
    if (relu) v = fmaxf(v, 0.0f);
    dst[idx] = v;
}

// ---------------------------------------------------------------------------
extern "C" void kernel_launch(void* const* d_in, const int* in_sizes, int n_in,
                              void* d_out, int out_size, void* d_ws, size_t ws_size,
                              hipStream_t stream) {
    constexpr int N = 50000, E = 400000, D = 128, L = 5;

    const int*   x     = (const int*)d_in[0];
    const int*   depth = (const int*)d_in[1];
    const int*   ei    = (const int*)d_in[2];
    const float* ea    = (const float*)d_in[3];
    const float* te    = (const float*)d_in[4];
    const float* ae    = (const float*)d_in[5];
    const float* de    = (const float*)d_in[6];
    const float* We    = (const float*)d_in[7];
    const float* be    = (const float*)d_in[8];
    const float* eps   = (const float*)d_in[9];
    const float* W1    = (const float*)d_in[10];
    const float* b1    = (const float*)d_in[11];
    const float* g1    = (const float*)d_in[12];
    const float* bt1   = (const float*)d_in[13];
    const float* W2    = (const float*)d_in[14];
    const float* b2    = (const float*)d_in[15];
    const float* go    = (const float*)d_in[16];
    const float* bo    = (const float*)d_in[17];
    float* out = (float*)d_out;

    char* wsp = (char*)d_ws;
    auto alloc = [&](size_t bytes) -> void* {
        void* p = wsp;
        wsp += (bytes + 255) & ~(size_t)255;
        return p;
    };
    float*     h     = (float*)alloc((size_t)N * D * 4);
    float*     agg   = (float*)alloc((size_t)N * D * 4);
    float*     t     = (float*)alloc((size_t)N * 2 * D * 4);
    float*     u     = (float*)alloc((size_t)N * D * 4);
    float*     stats = (float*)alloc(768 * 4);
    float*     ab1   = (float*)alloc(512 * 4);
    float*     ab2   = (float*)alloc(256 * 4);
    _Float16*  W1T   = (_Float16*)alloc(256 * 128 * 2);
    _Float16*  W2T   = (_Float16*)alloc(128 * 256 * 2);

    float* cs1 = stats;        float* cq1 = stats + 256;
    float* cs2 = stats + 512;  float* cq2 = stats + 640;

    node_encode<<<(N * 32 + 255) / 256, 256, 0, stream>>>(x, depth, te, ae, de, h, N);

    const dim3 gGrid1((N + 63) / 64, 2);
    const dim3 gGrid2((N + 63) / 64, 1);

    for (int l = 0; l < L; ++l) {
        zero_f32<<<(N * D + 255) / 256, 256, 0, stream>>>(agg, N * D);
        zero_f32<<<3, 256, 0, stream>>>(stats, 768);
        convert_wT<<<(128 * 256 + 255) / 256, 256, 0, stream>>>(W1 + (size_t)l * 128 * 256, W1T, 128, 256);
        convert_wT<<<(256 * 128 + 255) / 256, 256, 0, stream>>>(W2 + (size_t)l * 256 * 128, W2T, 256, 128);

        edge_scatter<<<(E * 32 + 255) / 256, 256, 0, stream>>>(
            h, ei, ea, We + (size_t)l * 256, be + (size_t)l * 128, agg, E);

        gemm_wmma<1, 128, 256><<<gGrid1, 128, 0, stream>>>(
            h, agg, eps + l, W1T, b1 + (size_t)l * 256, t, cs1, cq1, N);
        bn_finalize<<<1, 256, 0, stream>>>(cs1, cq1, g1 + (size_t)l * 256,
                                           bt1 + (size_t)l * 256, ab1, 256, 1.0f / N);

        gemm_wmma<2, 256, 128><<<gGrid2, 128, 0, stream>>>(
            t, ab1, nullptr, W2T, b2 + (size_t)l * 128, u, cs2, cq2, N);
        bn_finalize<<<1, 128, 0, stream>>>(cs2, cq2, go + (size_t)l * 128,
                                           bo + (size_t)l * 128, ab2, 128, 1.0f / N);

        float* dst = (l == L - 1) ? out : h;
        bn_apply<<<(N * D + 255) / 256, 256, 0, stream>>>(u, ab2, dst, N * D, 128, l < L - 1);
    }
}
